// Attention_46308337386245
// MI455X (gfx1250) — matrix operand order repaired and checked by reading, not compiled, not run on hardware
//
#include <hip/hip_runtime.h>
#include <hip/hip_bf16.h>

typedef __attribute__((ext_vector_type(16))) __bf16 bf16x16;
typedef __attribute__((ext_vector_type(8)))  __bf16 bf16x8;
typedef __attribute__((ext_vector_type(8)))  float  f32x8;
typedef __attribute__((ext_vector_type(4)))  unsigned int u32x4;
typedef __attribute__((ext_vector_type(4)))  int v4i;

typedef __attribute__((address_space(1))) v4i gv4i;   // global int4
typedef __attribute__((address_space(3))) v4i lv4i;   // LDS int4

union FragU { bf16x8 h[2]; bf16x16 v; };

// A-operand fragment (16x32 bf16, MxK): lane r<16 holds row r, K {0..7,16..23};
// lanes 16..31 hold K {8..15,24..31}. Two 16B loads 32B apart.
__device__ __forceinline__ bf16x16 frag_a_ld(const __bf16* row0, long strideElts, int lane) {
  const int r = lane & 15;
  const int off = (lane < 16) ? 0 : 8;
  const __bf16* p = row0 + (long)r * strideElts + off;
  FragU u;
  u.h[0] = *(const bf16x8*)(p);
  u.h[1] = *(const bf16x8*)(p + 16);
  return u.v;
}

// B-operand fragment (32x16 bf16, KxN) from K-major storage: lane n holds
// column n with K {0..7,16..23} (lanes 0-15) or K {8..15,24..31} (lanes 16-31).
__device__ __forceinline__ bf16x16 frag_b_ld(const __bf16* col0, long strideElts, int lane) {
  const int nn = lane & 15;
  const int off = (lane < 16) ? 0 : 8;
  const __bf16* p = col0 + (long)nn * strideElts + off;
  FragU u;
  u.h[0] = *(const bf16x8*)(p);
  u.h[1] = *(const bf16x8*)(p + 16);
  return u.v;
}

__device__ __forceinline__ f32x8 wmma_bf16(bf16x16 a, bf16x16 b, f32x8 c) {
  return __builtin_amdgcn_wmma_f32_16x16x32_bf16(false, a, false, b, (short)0, c, false, false);
}

// ---------------------------------------------------------------------------
// Generic bf16 GEMM: Cout[M][N] = A[M][K] @ Bw[N][K]^T  (both K-major)
// Block: 128x128 tile, 256 threads = 8 waves (2 M-waves x 4 N-waves),
// each wave computes 64x32 via 4x2 WMMA tiles, K-step 32.
// Double-buffered LDS staging via async global->LDS copies (ASYNCcnt) when
// the toolchain exposes the gfx1250 async builtins; sync copies otherwise.
// ---------------------------------------------------------------------------
template <typename OutT>
__global__ __launch_bounds__(256) void gemm_bf16_kernel(
    const __bf16* __restrict__ A, const __bf16* __restrict__ Bw,
    OutT* __restrict__ Cout, int M, int N, int K) {
  __shared__ __align__(16) __bf16 lA[2][128 * 40];  // 32 bf16 data + 8 pad per row
  __shared__ __align__(16) __bf16 lB[2][128 * 40];
  const int tid = threadIdx.x;
  const int lane = tid & 31;
  const int wave = tid >> 5;
  const int wm = wave & 1;
  const int wn = wave >> 1;
  const int m0 = blockIdx.y * 128;
  const int n0 = blockIdx.x * 128;

  // Stage one 128x32 A tile + 128x32 B tile into LDS buffer `buf`.
  auto stage = [&](int buf, int k0) {
#pragma unroll
    for (int i = 0; i < 2; ++i) {
      const int idx = tid + 256 * i;
      const int row = idx >> 2;
      const int ch = (idx & 3) * 8;
#if __has_builtin(__builtin_amdgcn_global_load_async_to_lds_b128)
      __builtin_amdgcn_global_load_async_to_lds_b128(
          (gv4i*)&A[(long)(m0 + row) * K + k0 + ch],
          (lv4i*)&lA[buf][row * 40 + ch], 0, 0);
      __builtin_amdgcn_global_load_async_to_lds_b128(
          (gv4i*)&Bw[(long)(n0 + row) * K + k0 + ch],
          (lv4i*)&lB[buf][row * 40 + ch], 0, 0);
#else
      *(u32x4*)&lA[buf][row * 40 + ch] = *(const u32x4*)&A[(long)(m0 + row) * K + k0 + ch];
      *(u32x4*)&lB[buf][row * 40 + ch] = *(const u32x4*)&Bw[(long)(n0 + row) * K + k0 + ch];
#endif
    }
  };

  f32x8 acc[4][2] = {};
  const int nsteps = K >> 5;
  stage(0, 0);
  for (int s = 0; s < nsteps; ++s) {
    asm volatile("s_wait_asynccnt 0" ::: "memory");  // my staged tile is in LDS
    __syncthreads();                                  // everyone's tile is in LDS
    const int buf = s & 1;
    if (s + 1 < nsteps) stage(buf ^ 1, (s + 1) * 32); // overlap next tile with WMMA
    if (s + 2 < nsteps) {                             // warm GL2 for the tile after
      __builtin_prefetch(&A[(long)(m0 + (tid >> 1)) * K + (s + 2) * 32], 0, 1);
      __builtin_prefetch(&Bw[(long)(n0 + (tid >> 1)) * K + (s + 2) * 32], 0, 1);
    }
    bf16x16 af[4], bfr[2];
#pragma unroll
    for (int i = 0; i < 4; ++i) af[i] = frag_a_ld(&lA[buf][(wm * 64 + i * 16) * 40], 40, lane);
#pragma unroll
    for (int j = 0; j < 2; ++j) bfr[j] = frag_b_ld(&lB[buf][(wn * 32 + j * 16) * 40], 40, lane);
#pragma unroll
    for (int i = 0; i < 4; ++i)
#pragma unroll
      for (int j = 0; j < 2; ++j) acc[i][j] = wmma_bf16(af[i], bfr[j], acc[i][j]);
  }
  const int hlf = lane >> 4, nn = lane & 15;
#pragma unroll
  for (int i = 0; i < 4; ++i)
#pragma unroll
    for (int j = 0; j < 2; ++j)
#pragma unroll
      for (int r = 0; r < 8; ++r) {
        const int row = m0 + wm * 64 + i * 16 + r + 8 * hlf;
        const int col = n0 + wn * 32 + j * 16 + nn;
        Cout[(long)row * N + col] = (OutT)acc[i][j][r];
      }
}

// ---------------------------------------------------------------------------
// fp32 -> bf16 conversion
// ---------------------------------------------------------------------------
__global__ __launch_bounds__(256) void cvt_kernel(const float* __restrict__ in,
                                                  __bf16* __restrict__ outp, long n) {
  const long i = (long)blockIdx.x * 256 + threadIdx.x;
  if (i < n) outp[i] = (__bf16)in[i];
}

// ---------------------------------------------------------------------------
// RoPE: in [b][t][heads*128] bf16 -> out [b][heads][t][128] bf16
// ---------------------------------------------------------------------------
__global__ __launch_bounds__(256) void rope_kernel(
    const __bf16* __restrict__ in, const float* __restrict__ fc,
    __bf16* __restrict__ outp, int heads) {
  long idx = (long)blockIdx.x * 256 + threadIdx.x;
  const int i = (int)(idx & 63); idx >>= 6;
  const int t = (int)(idx & 2047); idx >>= 11;
  const int hh = (int)(idx % heads);
  const int b = (int)(idx / heads);
  const long ibase = (((long)b * 2048 + t) * heads + hh) * 128 + 2 * i;
  const float a  = (float)in[ibase];
  const float bb = (float)in[ibase + 1];
  const float cs = fc[t * 128 + 2 * i];
  const float sn = fc[t * 128 + 2 * i + 1];
  const long o = ((long)(b * heads + hh) * 2048 + t) * 128 + 2 * i;
  outp[o]     = (__bf16)(a * cs - bb * sn);
  outp[o + 1] = (__bf16)(a * sn + bb * cs);
}

// ---------------------------------------------------------------------------
// V transpose: in [b][t][8*128] -> out [b][kvh][128][2048]  (V^T, t-contiguous)
// ---------------------------------------------------------------------------
__global__ __launch_bounds__(256) void transv_kernel(
    const __bf16* __restrict__ in, __bf16* __restrict__ outp) {
  long idx = (long)blockIdx.x * 256 + threadIdx.x;
  const int t = (int)(idx & 2047); idx >>= 11;
  const int d = (int)(idx & 127); idx >>= 7;
  const int kh = (int)(idx & 7);
  const int b = (int)(idx >> 3);
  outp[((long)(b * 8 + kh) * 128 + d) * 2048 + t] =
      in[(((long)b * 2048 + t) * 8 + kh) * 128 + d];
}

// ---------------------------------------------------------------------------
// Flash attention: one wave per 16-row query tile. Online softmax, causal.
// Q [b][32][2048][128], K [b][8][2048][128], V^T [b][8][128][2048], all bf16.
// O out: [b][t][32*128] bf16.
// ---------------------------------------------------------------------------
__global__ __launch_bounds__(256) void flash_kernel(
    const __bf16* __restrict__ Qm, const __bf16* __restrict__ Km,
    const __bf16* __restrict__ Vtm, __bf16* __restrict__ Om) {
  __shared__ __align__(16) __bf16 lP[8][16 * 40];  // per-wave 16x32 P transpose pad
  const int tid = threadIdx.x;
  const int lane = tid & 31;
  const int wave = tid >> 5;
  const int tile = blockIdx.x * 8 + wave;   // B*H*(T/16) = 8192 tiles total
  const int qt = tile & 127;
  const int bh = tile >> 7;
  const int h = bh & 31;
  const int b = bh >> 5;
  const int kvh = h >> 2;                   // G = 4
  const int q0 = qt * 16;
  const __bf16* Qb = Qm + (((long)(b * 32 + h)) * 2048 + q0) * 128;
  const __bf16* Kb = Km + ((long)(b * 8 + kvh)) * 2048 * 128;
  const __bf16* Vb = Vtm + ((long)(b * 8 + kvh)) * 128 * 2048;

  bf16x16 qf[4];
#pragma unroll
  for (int c = 0; c < 4; ++c) qf[c] = frag_a_ld(Qb + c * 32, 128, lane);

  f32x8 accO[8] = {};
  float mrun[8], lrun[8];
#pragma unroll
  for (int r = 0; r < 8; ++r) { mrun[r] = -3.0e38f; lrun[r] = 0.f; }
  const int hlf = lane >> 4, nn = lane & 15;
  const float scale = 0.08838834764831845f;   // 1/sqrt(128)
  __bf16* pw = lP[wave];

  for (int j = 0; j < q0 + 16; j += 32) {
    // S = Q @ K^T over HD=128 (4 K-chunks), two 16-col score tiles
    f32x8 s0 = {}, s1 = {};
#pragma unroll
    for (int c = 0; c < 4; ++c) {
      bf16x16 k0f = frag_b_ld(Kb + (long)j * 128 + c * 32, 128, lane);
      bf16x16 k1f = frag_b_ld(Kb + ((long)j + 16) * 128 + c * 32, 128, lane);
      s0 = wmma_bf16(qf[c], k0f, s0);
      s1 = wmma_bf16(qf[c], k1f, s1);
    }
    float sf[8];
#pragma unroll
    for (int r = 0; r < 8; ++r) {
      const int row = q0 + r + 8 * hlf;
      float a0 = s0[r] * scale;
      float a1 = s1[r] * scale;
      if (j + nn > row) a0 = -3.0e38f;          // causal mask
      if (j + 16 + nn > row) a1 = -3.0e38f;
      float mx = fmaxf(a0, a1);                 // row max across 16 lanes/half
      mx = fmaxf(mx, __shfl_xor(mx, 1, 16));
      mx = fmaxf(mx, __shfl_xor(mx, 2, 16));
      mx = fmaxf(mx, __shfl_xor(mx, 4, 16));
      mx = fmaxf(mx, __shfl_xor(mx, 8, 16));
      const float mnew = fmaxf(mrun[r], mx);
      const float f = __expf(mrun[r] - mnew);
      const float p0 = __expf(a0 - mnew);
      const float p1 = __expf(a1 - mnew);
      float rsum = p0 + p1;
      rsum += __shfl_xor(rsum, 1, 16);
      rsum += __shfl_xor(rsum, 2, 16);
      rsum += __shfl_xor(rsum, 4, 16);
      rsum += __shfl_xor(rsum, 8, 16);
      lrun[r] = lrun[r] * f + rsum;
      mrun[r] = mnew;
      sf[r] = f;
      const int rowl = r + 8 * hlf;             // P (C-layout) -> LDS row-major
      pw[rowl * 40 + nn] = (__bf16)p0;
      pw[rowl * 40 + 16 + nn] = (__bf16)p1;
    }
#pragma unroll
    for (int dt = 0; dt < 8; ++dt)
#pragma unroll
      for (int r = 0; r < 8; ++r) accO[dt][r] *= sf[r];
    asm volatile("s_wait_dscnt 0" ::: "memory");
    const bf16x16 pf = frag_a_ld(pw, 40, lane); // P as A-fragment (16x32)
#pragma unroll
    for (int dt = 0; dt < 8; ++dt) {            // O += P @ V  (V^T t-major)
      bf16x16 vf = frag_b_ld(Vb + (long)dt * 16 * 2048 + j, 2048, lane);
      accO[dt] = wmma_bf16(pf, vf, accO[dt]);
    }
  }
#pragma unroll
  for (int r = 0; r < 8; ++r) {
    const float inv = 1.0f / lrun[r];
    const int row = q0 + r + 8 * hlf;
#pragma unroll
    for (int dt = 0; dt < 8; ++dt) {
      const int d = dt * 16 + nn;
      Om[((long)b * 2048 + row) * 4096 + h * 128 + d] = (__bf16)(accO[dt][r] * inv);
    }
  }
}

// ---------------------------------------------------------------------------
extern "C" void kernel_launch(void* const* d_in, const int* /*in_sizes*/, int /*n_in*/,
                              void* d_out, int /*out_size*/, void* d_ws, size_t /*ws_size*/,
                              hipStream_t stream) {
  const float* x  = (const float*)d_in[0];
  const float* fc = (const float*)d_in[1];
  const float* wq = (const float*)d_in[2];
  const float* wk = (const float*)d_in[3];
  const float* wv = (const float*)d_in[4];
  const float* wo = (const float*)d_in[5];
  float* out = (float*)d_out;
  char* ws = (char*)d_ws;

  const long BT = 4096, Cc = 4096, QD = 4096, KD = 1024;
  size_t off = 0;
  auto alloc = [&](size_t bytes) {
    char* p = ws + off;
    off += (bytes + 255) & ~(size_t)255;
    return p;
  };
  __bf16* xb  = (__bf16*)alloc(2ul * BT * Cc);
  __bf16* wqb = (__bf16*)alloc(2ul * QD * Cc);
  __bf16* wkb = (__bf16*)alloc(2ul * KD * Cc);
  __bf16* wvb = (__bf16*)alloc(2ul * KD * Cc);
  __bf16* wob = (__bf16*)alloc(2ul * Cc * QD);
  __bf16* qtm = (__bf16*)alloc(2ul * BT * QD);
  __bf16* ktm = (__bf16*)alloc(2ul * BT * KD);
  __bf16* vtm = (__bf16*)alloc(2ul * BT * KD);
  __bf16* Qd  = (__bf16*)alloc(2ul * BT * QD);
  __bf16* Kd  = (__bf16*)alloc(2ul * BT * KD);
  __bf16* Vtd = (__bf16*)alloc(2ul * BT * KD);
  __bf16* Od  = (__bf16*)alloc(2ul * BT * QD);

  // fp32 -> bf16
  cvt_kernel<<<(unsigned)((BT * Cc) / 256), 256, 0, stream>>>(x, xb, BT * Cc);
  cvt_kernel<<<(unsigned)((QD * Cc) / 256), 256, 0, stream>>>(wq, wqb, QD * Cc);
  cvt_kernel<<<(unsigned)((KD * Cc) / 256), 256, 0, stream>>>(wk, wkb, KD * Cc);
  cvt_kernel<<<(unsigned)((KD * Cc) / 256), 256, 0, stream>>>(wv, wvb, KD * Cc);
  cvt_kernel<<<(unsigned)((Cc * QD) / 256), 256, 0, stream>>>(wo, wob, Cc * QD);

  // QKV projections (WMMA GEMM, async double-buffered staging)
  gemm_bf16_kernel<__bf16><<<dim3(QD / 128, BT / 128), 256, 0, stream>>>(xb, wqb, qtm, (int)BT, (int)QD, (int)Cc);
  gemm_bf16_kernel<__bf16><<<dim3(KD / 128, BT / 128), 256, 0, stream>>>(xb, wkb, ktm, (int)BT, (int)KD, (int)Cc);
  gemm_bf16_kernel<__bf16><<<dim3(KD / 128, BT / 128), 256, 0, stream>>>(xb, wvb, vtm, (int)BT, (int)KD, (int)Cc);

  // RoPE + layout
  rope_kernel<<<(unsigned)((2L * 32 * 2048 * 64) / 256), 256, 0, stream>>>(qtm, fc, Qd, 32);
  rope_kernel<<<(unsigned)((2L * 8 * 2048 * 64) / 256), 256, 0, stream>>>(ktm, fc, Kd, 8);
  transv_kernel<<<(unsigned)((2L * 8 * 128 * 2048) / 256), 256, 0, stream>>>(vtm, Vtd);

  // Flash attention (WMMA), 8192 query tiles / 8 waves per block
  flash_kernel<<<1024, 256, 0, stream>>>(Qd, Kd, Vtd, Od);

  // Output projection (float out -> d_out)
  gemm_bf16_kernel<float><<<dim3(Cc / 128, BT / 128), 256, 0, stream>>>(Od, wob, out, (int)BT, (int)Cc, (int)QD);
}
